// DAGraph_26310969655755
// MI455X (gfx1250) — matrix operand order repaired and checked by reading, not compiled, not run on hardware
//
#include <hip/hip_runtime.h>
#include <hip/hip_bf16.h>

#define DIM 128
#define LEAKY 0.2f

typedef __attribute__((ext_vector_type(16))) __bf16 v16bf;
typedef __attribute__((ext_vector_type(8)))  float  v8f;

// ---------------------------------------------------------------------------
// Zero-fill (float4 granularity). Needed because atomics accumulate and the
// harness does not re-poison between replays.
// ---------------------------------------------------------------------------
__global__ void zero_f4(float4* __restrict__ p, long n4) {
    long i = (long)blockIdx.x * blockDim.x + threadIdx.x;
    if (i < n4) {
        float4 z; z.x = 0.f; z.y = 0.f; z.z = 0.f; z.w = 0.f;
        p[i] = z;
    }
}

// ---------------------------------------------------------------------------
// SpMM: out[rows[e], :] += vals[e] * x[cols[e], :]
// One wave32 per edge; lane handles 4 consecutive columns (float4).
// Gathers hit L2 (items_emb is 51MB < 192MB L2); scatter via f32 atomics.
// ---------------------------------------------------------------------------
__global__ __launch_bounds__(256) void spmm_warp(
    const int*   __restrict__ rows,
    const int*   __restrict__ cols,
    const float* __restrict__ vals,
    const float* __restrict__ x,
    float*       __restrict__ out,
    int n_edges)
{
    int warp = (int)((blockIdx.x * (unsigned)blockDim.x + threadIdx.x) >> 5);
    int lane = threadIdx.x & 31;
    if (warp >= n_edges) return;
    int r = rows[warp];
    int c = cols[warp];
    float v = vals[warp];
    const float4 d = ((const float4*)(x + (size_t)c * DIM))[lane];
    float* o = out + (size_t)r * DIM + lane * 4;
    atomicAdd(o + 0, v * d.x);
    atomicAdd(o + 1, v * d.y);
    atomicAdd(o + 2, v * d.z);
    atomicAdd(o + 3, v * d.w);
}

// ---------------------------------------------------------------------------
// Gated GEMM + row-sum logits:
//   scores[i, rel] = sum_n LeakyReLU( ((emb ⊙ nb) @ W)[i, n] )
// The GEMM result is never materialized — it only feeds the attention logits,
// so bf16 WMMA (v_wmma_f32_16x16x32_bf16) is precision-appropriate.
//
// Wave32 per 16-row tile. 8 waves / block. W transposed into LDS as bf16:
// Wl[n*128 + k] so B-register K-pairs are contiguous 32-bit LDS loads.
//
// ISA register layouts (wave32):
//  A (16x32 bf16): lane l: m = l&15, kb = (l<16)?0:8;
//                  reg i holds K = kb + (i<4 ? 2i : 16+2(i-4)) and K+1.
//  B (32x16 bf16): lane l: n = l&15; reg j holds K = kb + (j<4 ? 2j : 16+2(j-4)) and K+1, as A.
//  D (16x16 f32):  lane l: n = l&15; reg i = row ( (l<16)?i : 8+i ).
// ---------------------------------------------------------------------------
__global__ __launch_bounds__(256) void gate_gemm_scores(
    const float* __restrict__ emb,
    const float* __restrict__ nb,
    const float* __restrict__ W,      // [128][128] row-major f32 (K x N)
    float*       __restrict__ scores, // [n_items][3]
    int n_items, int rel)
{
    __shared__ __bf16 Wl[DIM * DIM];  // 32 KB, transposed: Wl[n*128+k] = W[k*128+n]

    for (int t = threadIdx.x; t < DIM * DIM; t += blockDim.x) {
        int k = t >> 7;          // row of W  (coalesced read across n)
        int n = t & 127;
        Wl[n * DIM + k] = (__bf16)W[t];
    }
    __syncthreads();

    const int wave = threadIdx.x >> 5;
    const int lane = threadIdx.x & 31;
    const int tile = blockIdx.x * 8 + wave;          // 16 rows per tile
    if (tile * 16 >= n_items) return;                // uniform per wave -> EXEC stays full

    const int half = lane >> 4;                      // 0 | 1
    const int lm   = lane & 15;

    // ---- build A tiles: gate = emb ⊙ nb, converted to bf16, A-layout ----
    int rowA = tile * 16 + lm;
    if (rowA >= n_items) rowA = n_items - 1;         // safe clamp (n_items % 16 == 0 anyway)
    const float* eRow = emb + (size_t)rowA * DIM;
    const float* gRow = nb  + (size_t)rowA * DIM;

    v16bf A[4];
    #pragma unroll
    for (int c = 0; c < 4; ++c) {
        #pragma unroll
        for (int i = 0; i < 8; ++i) {
            const int kl = (i < 4) ? (half * 8 + 2 * i) : (half * 8 + 16 + 2 * (i - 4));
            const int k  = c * 32 + kl;
            const float2 e = *(const float2*)(eRow + k);
            const float2 g = *(const float2*)(gRow + k);
            A[c][2 * i]     = (__bf16)(e.x * g.x);
            A[c][2 * i + 1] = (__bf16)(e.y * g.y);
        }
    }

    float rs[8];
    #pragma unroll
    for (int i = 0; i < 8; ++i) rs[i] = 0.f;

    const int kbB = half * 16;
    #pragma unroll
    for (int nt = 0; nt < 8; ++nt) {
        v8f acc = {};
        const __bf16* wb = &Wl[(nt * 16 + lm) * DIM];
        #pragma unroll
        for (int c = 0; c < 4; ++c) {
            v16bf B;
            #pragma unroll
            for (int j = 0; j < 8; ++j) {
                const int k = c * 32 + ((j < 4) ? (half * 8 + 2 * j) : (half * 8 + 16 + 2 * (j - 4)));   // contiguous pair in LDS
                B[2 * j]     = wb[k];
                B[2 * j + 1] = wb[k + 1];
            }
            acc = __builtin_amdgcn_wmma_f32_16x16x32_bf16(
                      false, A[c], false, B, (short)0, acc, false, false);
        }
        #pragma unroll
        for (int i = 0; i < 8; ++i) {
            const float v = acc[i];
            rs[i] += (v > 0.f) ? v : LEAKY * v;
        }
    }

    // reduce over the 16 lanes sharing each half (sum over N)
    #pragma unroll
    for (int m = 1; m <= 8; m <<= 1) {
        #pragma unroll
        for (int i = 0; i < 8; ++i)
            rs[i] += __shfl_xor(rs[i], m, 32);
    }

    if (lm == 0) {
        const int mbase = tile * 16 + half * 8;
        #pragma unroll
        for (int i = 0; i < 8; ++i)
            if (mbase + i < n_items)
                scores[(size_t)(mbase + i) * 3 + rel] = rs[i];
    }
}

// ---------------------------------------------------------------------------
// Relation-attention softmax + weighted blend of the three NB matrices.
// One thread per (item, 4-column chunk).
// ---------------------------------------------------------------------------
__global__ __launch_bounds__(256) void combine_kernel(
    const float* __restrict__ nb0,
    const float* __restrict__ nb1,
    const float* __restrict__ nb2,
    const float* __restrict__ scores,
    float*       __restrict__ out_i,
    int n_items)
{
    const float inv_sqrt_d = 0.08838834764831845f;   // 1/sqrt(128)
    long idx = (long)blockIdx.x * blockDim.x + threadIdx.x;
    long total = (long)n_items * (DIM / 4);
    if (idx >= total) return;
    const int i = (int)(idx >> 5);                   // 32 float4 chunks per row
    const int cv = (int)(idx & 31);

    float s0 = scores[(size_t)i * 3 + 0] * inv_sqrt_d;
    float s1 = scores[(size_t)i * 3 + 1] * inv_sqrt_d;
    float s2 = scores[(size_t)i * 3 + 2] * inv_sqrt_d;
    float mx = fmaxf(s0, fmaxf(s1, s2));
    float e0 = __expf(s0 - mx), e1 = __expf(s1 - mx), e2 = __expf(s2 - mx);
    float inv = 1.0f / (e0 + e1 + e2);
    float w0 = e0 * inv, w1 = e1 * inv, w2 = e2 * inv;

    const size_t off = (size_t)i * DIM + (size_t)cv * 4;
    const float4 a = *(const float4*)(nb0 + off);
    const float4 b = *(const float4*)(nb1 + off);
    const float4 c = *(const float4*)(nb2 + off);
    float4 o;
    o.x = w0 * a.x + w1 * b.x + w2 * c.x;
    o.y = w0 * a.y + w1 * b.y + w2 * c.y;
    o.z = w0 * a.z + w1 * b.z + w2 * c.z;
    o.w = w0 * a.w + w1 * b.w + w2 * c.w;
    *(float4*)(out_i + off) = o;
}

// ---------------------------------------------------------------------------
// Launch
// ---------------------------------------------------------------------------
extern "C" void kernel_launch(void* const* d_in, const int* in_sizes, int n_in,
                              void* d_out, int out_size, void* d_ws, size_t ws_size,
                              hipStream_t stream) {
    // input order per setup_inputs():
    //  0 b, 1 n_users, 2 items_emb, 3-5 e2e_in (rows,cols,val),
    //  6-8 e2e_out, 9-11 e2p_out, 12-14 ui, 15-17 weight matrices
    const float* items_emb = (const float*)d_in[2];
    const int*   r0 = (const int*)d_in[3];
    const int*   c0 = (const int*)d_in[4];
    const float* v0 = (const float*)d_in[5];
    const int*   r1 = (const int*)d_in[6];
    const int*   c1 = (const int*)d_in[7];
    const float* v1 = (const float*)d_in[8];
    const int*   r2 = (const int*)d_in[9];
    const int*   c2 = (const int*)d_in[10];
    const float* v2 = (const float*)d_in[11];
    const int*   ru = (const int*)d_in[12];
    const int*   cu = (const int*)d_in[13];
    const float* vu = (const float*)d_in[14];
    const float* W0 = (const float*)d_in[15];
    const float* W1 = (const float*)d_in[16];
    const float* W2 = (const float*)d_in[17];

    const int n_items = in_sizes[2] / DIM;
    const int n_users = out_size / DIM - n_items;   // out = [users*D | items*D]
    const int E0 = in_sizes[3];
    const int E1 = in_sizes[6];
    const int E2 = in_sizes[9];
    const int Eu = in_sizes[12];

    // workspace layout (floats): nb0 | nb1 | nb2 | scores
    const size_t nbElems = (size_t)n_items * DIM;
    float* nb0    = (float*)d_ws;
    float* nb1    = nb0 + nbElems;
    float* nb2    = nb1 + nbElems;
    float* scores = nb2 + nbElems;

    float* out_u = (float*)d_out;
    float* out_i = out_u + (size_t)n_users * DIM;

    // 1) zero accumulators: 3 NB buffers (contiguous) and the user-output half
    {
        long n4 = (long)(3 * nbElems) / 4;
        zero_f4<<<(unsigned)((n4 + 255) / 256), 256, 0, stream>>>((float4*)nb0, n4);
        long u4 = ((long)n_users * DIM) / 4;
        zero_f4<<<(unsigned)((u4 + 255) / 256), 256, 0, stream>>>((float4*)out_u, u4);
    }

    // 2) SpMMs (wave per edge, 8 edges per 256-thread block)
    spmm_warp<<<(unsigned)((E0 + 7) / 8), 256, 0, stream>>>(r0, c0, v0, items_emb, nb0, E0);
    spmm_warp<<<(unsigned)((E1 + 7) / 8), 256, 0, stream>>>(r1, c1, v1, items_emb, nb1, E1);
    spmm_warp<<<(unsigned)((E2 + 7) / 8), 256, 0, stream>>>(r2, c2, v2, items_emb, nb2, E2);
    spmm_warp<<<(unsigned)((Eu + 7) / 8), 256, 0, stream>>>(ru, cu, vu, items_emb, out_u, Eu);

    // 3) gated GEMM -> attention logits via bf16 WMMA
    {
        const int tiles  = (n_items + 15) / 16;
        const unsigned blocks = (unsigned)((tiles + 7) / 8);
        gate_gemm_scores<<<blocks, 256, 0, stream>>>(items_emb, nb0, W0, scores, n_items, 0);
        gate_gemm_scores<<<blocks, 256, 0, stream>>>(items_emb, nb1, W1, scores, n_items, 1);
        gate_gemm_scores<<<blocks, 256, 0, stream>>>(items_emb, nb2, W2, scores, n_items, 2);
    }

    // 4) softmax + weighted combine into item-output half
    {
        long total = (long)n_items * (DIM / 4);
        combine_kernel<<<(unsigned)((total + 255) / 256), 256, 0, stream>>>(
            nb0, nb1, nb2, scores, out_i, n_items);
    }
}
